// MPML_57509612093615
// MI455X (gfx1250) — compile-verified, matrix operand order checked by reading, not run on hardware
//
#include <hip/hip_runtime.h>
#include <math.h>

#define B_    64
#define T_    10
#define SEG_  1184
#define SIG_  8
#define ADJ_  64
#define HID_  1024
#define NCLS_ 5
#define NS_   640          // B*T
#define L1_   1153
#define L1P_  287
#define L2_   272
#define L2P_  135
#define L3_   128
#define TOPK_ 1638

typedef __attribute__((ext_vector_type(16))) _Float16 v16h;
typedef __attribute__((ext_vector_type(8)))  _Float16 v8h;
typedef __attribute__((ext_vector_type(8)))  float    v8f;

// ---------------------------------------------------------------------------
// CDNA5 async global->LDS copy (ASYNCcnt path, ISA 10.7/15.18).
// Generic LDS pointer low 32 bits == LDS byte offset (ISA 10.2 aperture rules),
// so we can form the VDST operand by truncation. GV mode: 64-bit vaddr, no saddr.
// ---------------------------------------------------------------------------
__device__ __forceinline__ void async_copy_b128(void* lds_dst, const void* gsrc) {
  unsigned l = (unsigned)(uintptr_t)lds_dst;
  unsigned long long g = (unsigned long long)(uintptr_t)gsrc;
  asm volatile("global_load_async_to_lds_b128 %0, %1, off"
               :
               : "v"(l), "v"(g)
               : "memory");
}
__device__ __forceinline__ void wait_async0() {
  asm volatile("s_wait_asynccnt 0x0" ::: "memory");
}

// ---------------------------------------------------------------------------
// WMMA fragment helpers (wave32, CDNA5 layouts per ISA 7.12.2):
//  A 16x32 f16 : lane l (0..15) row M=l, lane l+16 same rows; lane-group g=l>>4
//                holds K = {g*8..g*8+7} U {g*8+16..g*8+23}  (contiguous pairs)
//  B 32x16 f16 : symmetric, lane holds column N=l&15, same K pattern
//  C/D 16x16 f32: VGPR r -> (M = r + (l>>4)*8, N = l&15)
// ---------------------------------------------------------------------------
__device__ __forceinline__ v16h pack2(const _Float16* p) {
  v8h lo = *(const v8h*)(p);
  v8h hi = *(const v8h*)(p + 16);
  v16h r;
#pragma unroll
  for (int e = 0; e < 8; ++e) { r[e] = lo[e]; r[e + 8] = hi[e]; }
  return r;
}

// A tile from row-major (M x K), leading dim lda (in halves)
__device__ __forceinline__ v16h load_a(const _Float16* src, int lda, int row0, int k0) {
  int lane = threadIdx.x & 31;
  const _Float16* p = src + (size_t)(row0 + (lane & 15)) * lda + k0 + ((lane >> 4) << 3);
  return pack2(p);
}

// B tile where logical B(k,n) = W[n*ldw + k]  (i.e. B = W^T, W row-major (N x K))
__device__ __forceinline__ v16h load_b_tr(const _Float16* w, int ldw, int k0, int col0) {
  int lane = threadIdx.x & 31;
  const _Float16* p = w + (size_t)(col0 + (lane & 15)) * ldw + k0 + ((lane >> 4) << 3);
  return pack2(p);
}

// B tile from row-major (K x N), leading dim ldb
__device__ __forceinline__ v16h load_b_rm(const _Float16* src, int ldb, int k0, int col0) {
  int lane = threadIdx.x & 31;
  int c = col0 + (lane & 15);
  int g = (lane >> 4) << 3;
  v16h b;
#pragma unroll
  for (int e = 0; e < 8; ++e) {
    b[e]     = src[(size_t)(k0 + g + e) * ldb + c];
    b[e + 8] = src[(size_t)(k0 + g + e + 16) * ldb + c];
  }
  return b;
}

__device__ __forceinline__ v8f wmma32(v16h a, v16h b, v8f c) {
  return __builtin_amdgcn_wmma_f32_16x16x32_f16(false, a, false, b, (short)0, c,
                                                false, false);
}

__device__ __forceinline__ float sigm(float x) { return 1.0f / (1.0f + expf(-x)); }

// ---------------------------------------------------------------------------
// Utility kernels
// ---------------------------------------------------------------------------
__global__ void k_f32h(const float* __restrict__ s, _Float16* __restrict__ d, int n) {
  int i = blockIdx.x * 256 + threadIdx.x;
  if (i < n) d[i] = (_Float16)s[i];
}

__global__ void k_zero(unsigned* __restrict__ p, int n) {
  int i = blockIdx.x * 256 + threadIdx.x;
  if (i < n) p[i] = 0u;
}

// ---------------------------------------------------------------------------
// Conv stack
// ---------------------------------------------------------------------------
__global__ void k_conv1(const float* __restrict__ x, const float* __restrict__ w,
                        const float* __restrict__ bias, float* __restrict__ out) {
  int idx = blockIdx.x * 256 + threadIdx.x;
  if (idx >= NS_ * 16 * L1_) return;
  int l = idx % L1_;
  int c = (idx / L1_) & 15;
  int n = idx / (L1_ * 16);
  float acc = bias[c];
  const float* xp = x + ((size_t)n * SEG_ + l) * SIG_;   // channel-last input
  const float* wp = w + (size_t)c * (SIG_ * 32);         // w[c][s][k]
  for (int k = 0; k < 32; ++k)
    for (int s = 0; s < 8; ++s)
      acc += xp[k * 8 + s] * wp[s * 32 + k];
  out[idx] = acc;
}

__global__ void k_conv(const float* __restrict__ in, const float* __restrict__ w,
                       const float* __restrict__ bias, float* __restrict__ out,
                       int CO, int CI, int KW, int Lin, int Lout, int total) {
  int idx = blockIdx.x * 256 + threadIdx.x;
  if (idx >= total) return;
  int l = idx % Lout;
  int c = (idx / Lout) % CO;
  int n = idx / (Lout * CO);
  float acc = bias[c];
  const float* ip = in + (size_t)n * CI * Lin + l;
  const float* wp = w + (size_t)c * CI * KW;
  for (int ci = 0; ci < CI; ++ci)
    for (int k = 0; k < KW; ++k)
      acc += ip[(size_t)ci * Lin + k] * wp[ci * KW + k];
  out[idx] = acc;
}

__global__ void k_bnstats(const float* __restrict__ buf, int C, int NL, int L,
                          const float* __restrict__ g, const float* __restrict__ b,
                          float* __restrict__ scale, float* __restrict__ shift) {
  __shared__ float ss[256], sq[256];
  int c = blockIdx.x, tid = threadIdx.x;
  float s = 0.f, q = 0.f;
  for (int i = tid; i < NL; i += 256) {
    int n = i / L, l = i - n * L;
    float v = buf[((size_t)n * C + c) * L + l];
    s += v; q += v * v;
  }
  ss[tid] = s; sq[tid] = q; __syncthreads();
  for (int o = 128; o > 0; o >>= 1) {
    if (tid < o) { ss[tid] += ss[tid + o]; sq[tid] += sq[tid + o]; }
    __syncthreads();
  }
  if (tid == 0) {
    float mean = ss[0] / (float)NL;
    float var  = sq[0] / (float)NL - mean * mean;
    float sc = g[c] * rsqrtf(var + 1e-5f);
    scale[c] = sc; shift[c] = b[c] - mean * sc;
  }
}

__global__ void k_bnpool(const float* __restrict__ in, float* __restrict__ out,
                         int C, int Lin, int Lout, int win, int stride,
                         const float* __restrict__ scale, const float* __restrict__ shift,
                         int total) {
  int idx = blockIdx.x * 256 + threadIdx.x;
  if (idx >= total) return;
  int l = idx % Lout;
  int c = (idx / Lout) % C;
  int n = idx / (Lout * C);
  float sc = scale[c], sh = shift[c];
  const float* p = in + ((size_t)n * C + c) * Lin + l * stride;
  float m = -1e30f;
  for (int w = 0; w < win; ++w) m = fmaxf(m, p[w] * sc + sh);
  out[idx] = fmaxf(m, 0.f);   // relu then maxpool == maxpool then relu
}

// BN+relu on conv3 and pairwise max -> f1 (N,64,64) f32 + f16 copy
__global__ void k_f1(const float* __restrict__ c3, const float* __restrict__ scale,
                     const float* __restrict__ shift, float* __restrict__ f1,
                     _Float16* __restrict__ f1h) {
  int idx = blockIdx.x * 256 + threadIdx.x;
  if (idx >= NS_ * ADJ_ * ADJ_) return;
  int a = idx & 63;
  int c = (idx >> 6) & 63;
  int n = idx >> 12;
  const float* p = c3 + ((size_t)n * 64 + c) * L3_ + 2 * a;
  float sc = scale[c], sh = shift[c];
  float v = fmaxf(p[0] * sc + sh, p[1] * sc + sh);
  v = fmaxf(v, 0.f);
  f1[idx] = v;
  f1h[idx] = (_Float16)v;
}

// ---------------------------------------------------------------------------
// Fused per-sample attention + Gaussian adjacency -> adj (pre top-k)
// ---------------------------------------------------------------------------
__device__ __forceinline__ void softmax_row64(float* row) {
  float m = row[0];
  for (int i = 1; i < 64; ++i) m = fmaxf(m, row[i]);
  float s = 0.f;
  for (int i = 0; i < 64; ++i) { float e = expf(row[i] - m); row[i] = e; s += e; }
  float inv = 1.f / s;
  for (int i = 0; i < 64; ++i) row[i] *= inv;
}

__global__ void k_attention(const float* __restrict__ f1, const _Float16* __restrict__ f1h,
                            const _Float16* __restrict__ qw, const float* __restrict__ qb,
                            const _Float16* __restrict__ kw, const float* __restrict__ kb,
                            const _Float16* __restrict__ vw, const float* __restrict__ vb,
                            const float* __restrict__ u, float* __restrict__ adj) {
  __shared__ __align__(16) unsigned char smem[49664];
  _Float16* s_f1h = (_Float16*)smem;            // 4096 halves
  _Float16* s_Qh  = s_f1h + 4096;               // Q -> later att(f16)
  _Float16* s_KTh = s_Qh + 4096;                // K^T
  _Float16* s_Vh  = s_KTh + 4096;               // V
  float* s_F  = (float*)(smem + 32768);         // att f32 -> adj_att f32
  float* s_G  = (float*)(s_Qh);                 // adj_g f32 (overlays Qh+KTh late)
  float* s_r2 = (float*)(smem + 49152);
  float* s_c2 = s_r2 + 64;

  int tid = threadIdx.x, wave = tid >> 5, lane = tid & 31;
  int n = blockIdx.x;
  const size_t base = (size_t)n * 4096;

  // async global->LDS staging of f1 (f16): 512 x 16B chunks, 2 per thread
  for (int i = tid * 8; i < 4096; i += 256 * 8)
    async_copy_b128(s_f1h + i, f1h + base + i);
  if (tid < 64) {
    float s = 0.f;
    for (int d = 0; d < 64; ++d) { float v = f1[base + tid * 64 + d]; s += v * v; }
    s_r2[tid] = s;
  } else if (tid < 128) {
    int j = tid - 64; float s = 0.f;
    for (int h = 0; h < 64; ++h) { float v = f1[base + h * 64 + j]; s += v * v; }
    s_c2[j] = s;
  }
  wait_async0();
  __syncthreads();

  // Q = f1 qw^T + qb ; K^T stored transposed ; V = f1 vw^T + vb
  for (int t2 = 0; t2 < 2; ++t2) {
    int tile = wave + (t2 << 3);
    int r0 = (tile >> 2) << 4, c0 = (tile & 3) << 4;
    v8f aq = {}, ak = {}, av = {};
    for (int k0 = 0; k0 < 64; k0 += 32) {
      v16h a = load_a(s_f1h, 64, r0, k0);
      aq = wmma32(a, load_b_tr(qw, 64, k0, c0), aq);
      ak = wmma32(a, load_b_tr(kw, 64, k0, c0), ak);
      av = wmma32(a, load_b_tr(vw, 64, k0, c0), av);
    }
    int col = c0 + (lane & 15), rb = r0 + ((lane >> 4) << 3);
#pragma unroll
    for (int r = 0; r < 8; ++r) {
      int row = rb + r;
      s_Qh[row * 64 + col]  = (_Float16)(aq[r] + qb[col]);
      s_KTh[col * 64 + row] = (_Float16)(ak[r] + kb[col]);
      s_Vh[row * 64 + col]  = (_Float16)(av[r] + vb[col]);
    }
  }
  __syncthreads();

  // att(w,v) = sum_h K[h,w] Q[h,v] / 8  ==  K^T @ Q
  for (int t2 = 0; t2 < 2; ++t2) {
    int tile = wave + (t2 << 3);
    int r0 = (tile >> 2) << 4, c0 = (tile & 3) << 4;
    v8f acc = {};
    for (int k0 = 0; k0 < 64; k0 += 32)
      acc = wmma32(load_a(s_KTh, 64, r0, k0), load_b_rm(s_Qh, 64, k0, c0), acc);
    int col = c0 + (lane & 15), rb = r0 + ((lane >> 4) << 3);
#pragma unroll
    for (int r = 0; r < 8; ++r) s_F[(rb + r) * 64 + col] = acc[r] * 0.125f;
  }
  __syncthreads();
  if (tid < 64) softmax_row64(s_F + tid * 64);
  __syncthreads();
  for (int i = tid; i < 4096; i += 256) s_Qh[i] = (_Float16)s_F[i];  // att f16 (Q dead)
  __syncthreads();

  // adj_att = V @ att
  for (int t2 = 0; t2 < 2; ++t2) {
    int tile = wave + (t2 << 3);
    int r0 = (tile >> 2) << 4, c0 = (tile & 3) << 4;
    v8f acc = {};
    for (int k0 = 0; k0 < 64; k0 += 32)
      acc = wmma32(load_a(s_Vh, 64, r0, k0), load_b_rm(s_Qh, 64, k0, c0), acc);
    int col = c0 + (lane & 15), rb = r0 + ((lane >> 4) << 3);
#pragma unroll
    for (int r = 0; r < 8; ++r) s_F[(rb + r) * 64 + col] = acc[r];
  }
  __syncthreads();

  // cross = f1 @ f1 ; adj_g = exp(-(r2+c2-2*cross)/2)   (Qh/KTh dead -> s_G)
  for (int t2 = 0; t2 < 2; ++t2) {
    int tile = wave + (t2 << 3);
    int r0 = (tile >> 2) << 4, c0 = (tile & 3) << 4;
    v8f acc = {};
    for (int k0 = 0; k0 < 64; k0 += 32)
      acc = wmma32(load_a(s_f1h, 64, r0, k0), load_b_rm(s_f1h, 64, k0, c0), acc);
    int col = c0 + (lane & 15), rb = r0 + ((lane >> 4) << 3);
#pragma unroll
    for (int r = 0; r < 8; ++r) {
      int row = rb + r;
      float d = s_r2[row] + s_c2[col] - 2.f * acc[r];
      s_G[row * 64 + col] = expf(-0.5f * d);
    }
  }
  __syncthreads();
  if (tid < 64) softmax_row64(s_F + tid * 64);
  else if (tid < 128) softmax_row64(s_G + (tid - 64) * 64);
  __syncthreads();

  for (int i = tid; i < 4096; i += 256) {
    float uu = u[base + i];
    float gum = -logf(-logf(uu + 1e-20f) + 1e-20f);
    adj[base + i] = s_F[i] + s_G[i] + gum;
  }
}

// ---------------------------------------------------------------------------
// Exact top-k threshold via bitonic sort, then degree -> dinv
// ---------------------------------------------------------------------------
__global__ void k_topk(const float* __restrict__ adj, float* __restrict__ dinv) {
  __shared__ __align__(16) float s[4096];
  __shared__ float s_thr;
  int n = blockIdx.x, tid = threadIdx.x;
  const size_t base = (size_t)n * 4096;
  // async global->LDS staging: 1024 x 16B chunks, 4 per thread
  for (int i = tid * 4; i < 4096; i += 256 * 4)
    async_copy_b128(s + i, adj + base + i);
  wait_async0();
  __syncthreads();
  for (int k = 2; k <= 4096; k <<= 1)
    for (int j = k >> 1; j > 0; j >>= 1) {
      for (int t = tid; t < 4096; t += 256) {
        int ixj = t ^ j;
        if (ixj > t) {
          float a = s[t], b = s[ixj];
          bool up = ((t & k) == 0);
          if (up ? (a > b) : (a < b)) { s[t] = b; s[ixj] = a; }
        }
      }
      __syncthreads();
    }
  if (tid == 0) s_thr = s[4096 - TOPK_];   // k-th largest
  __syncthreads();
  if (tid < 64) {
    float thr = s_thr, deg = 1.f;          // diagonal forced to 1
    for (int i = 0; i < 64; ++i)
      if (i != tid && adj[base + i * 64 + tid] >= thr) deg += 1.f;
    dinv[n * 64 + tid] = rsqrtf(deg);
  }
}

// ---------------------------------------------------------------------------
// GCN layer: out = relu( (dinv[i] * x) @ W + b )
// ---------------------------------------------------------------------------
__global__ void k_gcn(const float* __restrict__ x, const float* __restrict__ dinv,
                      const _Float16* __restrict__ wh, const float* __restrict__ bias,
                      float* __restrict__ outf, _Float16* __restrict__ outh) {
  __shared__ _Float16 s_xh[4096];
  int tid = threadIdx.x, wave = tid >> 5, lane = tid & 31;
  int n = blockIdx.x;
  const size_t base = (size_t)n * 4096;
  for (int i = tid; i < 4096; i += 256) {
    int row = i >> 6;
    s_xh[i] = (_Float16)(x[base + i] * dinv[n * 64 + row]);
  }
  __syncthreads();
  for (int t2 = 0; t2 < 2; ++t2) {
    int tile = wave + (t2 << 3);
    int r0 = (tile >> 2) << 4, c0 = (tile & 3) << 4;
    v8f acc = {};
    for (int k0 = 0; k0 < 64; k0 += 32)
      acc = wmma32(load_a(s_xh, 64, r0, k0), load_b_rm(wh, 64, k0, c0), acc);
    int col = c0 + (lane & 15), rb = r0 + ((lane >> 4) << 3);
#pragma unroll
    for (int r = 0; r < 8; ++r) {
      float v = fmaxf(acc[r] + bias[col], 0.f);
      size_t o = base + (size_t)(rb + r) * 64 + col;
      if (outf) outf[o] = v;
      if (outh) outh[o] = (_Float16)v;
    }
  }
}

// ---------------------------------------------------------------------------
// LSTM step GEMM: gates(64x4096) = x@Wih^T + h@Whh^T   (no bias)
// 1024 wave-tiles of 16x16; grid 128 x 256 threads (8 waves)
// ---------------------------------------------------------------------------
__global__ void k_lstm_gemm(const _Float16* __restrict__ xh, int ldx, int Kx,
                            const _Float16* __restrict__ hh,
                            const _Float16* __restrict__ wih,
                            const _Float16* __restrict__ whh,
                            float* __restrict__ gates) {
  int wave = threadIdx.x >> 5, lane = threadIdx.x & 31;
  int gt = blockIdx.x * 8 + wave;
  int m0 = (gt >> 8) << 4;
  int n0 = (gt & 255) << 4;
  v8f acc = {};
  for (int k0 = 0; k0 < Kx; k0 += 32) {
    if (k0 + 32 < Kx)
      __builtin_prefetch(wih + (size_t)(n0 + (lane & 15)) * Kx + k0 + 32, 0, 1);
    acc = wmma32(load_a(xh, ldx, m0, k0), load_b_tr(wih, Kx, k0, n0), acc);
  }
  for (int k0 = 0; k0 < HID_; k0 += 32)
    acc = wmma32(load_a(hh, HID_, m0, k0), load_b_tr(whh, HID_, k0, n0), acc);
  int col = n0 + (lane & 15), rb = m0 + ((lane >> 4) << 3);
#pragma unroll
  for (int r = 0; r < 8; ++r) gates[(size_t)(rb + r) * 4096 + col] = acc[r];
}

__global__ void k_lstm_cell(const float* __restrict__ gates,
                            const float* __restrict__ bih, const float* __restrict__ bhh,
                            float* __restrict__ cst, _Float16* __restrict__ hhst,
                            float* __restrict__ hf, _Float16* __restrict__ yh, int t) {
  int idx = blockIdx.x * 256 + threadIdx.x;
  if (idx >= B_ * HID_) return;
  int b = idx >> 10, j = idx & 1023;
  const float* g = gates + (size_t)b * 4096;
  float ig = g[j]        + bih[j]        + bhh[j];
  float fg = g[1024 + j] + bih[1024 + j] + bhh[1024 + j];
  float gg = g[2048 + j] + bih[2048 + j] + bhh[2048 + j];
  float og = g[3072 + j] + bih[3072 + j] + bhh[3072 + j];
  float c = sigm(fg) * cst[idx] + sigm(ig) * tanhf(gg);
  float h = sigm(og) * tanhf(c);
  cst[idx] = c;
  hhst[idx] = (_Float16)h;
  hf[idx] = h;
  if (yh) yh[((size_t)b * T_ + t) * HID_ + j] = (_Float16)h;
}

__global__ void k_cls(const float* __restrict__ f4, const float* __restrict__ w,
                      const float* __restrict__ b, float* __restrict__ out) {
  int t = threadIdx.x;
  if (t >= B_ * NCLS_) return;
  int bb = t / NCLS_, c = t % NCLS_;
  float acc = b[c];
  for (int j = 0; j < HID_; ++j) acc += f4[bb * HID_ + j] * w[c * HID_ + j];
  out[t] = acc;
}

// ---------------------------------------------------------------------------
extern "C" void kernel_launch(void* const* d_in, const int* in_sizes, int n_in,
                              void* d_out, int out_size, void* d_ws, size_t ws_size,
                              hipStream_t stream) {
  (void)in_sizes; (void)n_in; (void)out_size; (void)ws_size;
  const float* data   = (const float*)d_in[0];
  const float* u      = (const float*)d_in[1];
  const float* c1w    = (const float*)d_in[2];
  const float* c1b    = (const float*)d_in[3];
  const float* bn1g   = (const float*)d_in[4];
  const float* bn1b   = (const float*)d_in[5];
  const float* c2w    = (const float*)d_in[6];
  const float* c2b    = (const float*)d_in[7];
  const float* bn2g   = (const float*)d_in[8];
  const float* bn2b   = (const float*)d_in[9];
  const float* c3w    = (const float*)d_in[10];
  const float* c3b    = (const float*)d_in[11];
  const float* bn3g   = (const float*)d_in[12];
  const float* bn3b   = (const float*)d_in[13];
  const float* q_w    = (const float*)d_in[14];
  const float* q_b    = (const float*)d_in[15];
  const float* k_w    = (const float*)d_in[16];
  const float* k_b    = (const float*)d_in[17];
  const float* v_w    = (const float*)d_in[18];
  const float* v_b    = (const float*)d_in[19];
  const float* g1w    = (const float*)d_in[20];
  const float* g1b    = (const float*)d_in[21];
  const float* g2w    = (const float*)d_in[22];
  const float* g2b    = (const float*)d_in[23];
  const float* wih0   = (const float*)d_in[24];
  const float* whh0   = (const float*)d_in[25];
  const float* bih0   = (const float*)d_in[26];
  const float* bhh0   = (const float*)d_in[27];
  const float* wih1   = (const float*)d_in[28];
  const float* whh1   = (const float*)d_in[29];
  const float* bih1   = (const float*)d_in[30];
  const float* bhh1   = (const float*)d_in[31];
  const float* clsw   = (const float*)d_in[32];
  const float* clsb   = (const float*)d_in[33];
  float* out = (float*)d_out;

  char* w8 = (char*)d_ws;
  size_t off = 0;
  auto A = [&](size_t bytes) -> char* {
    char* p = w8 + off;
    off += (bytes + 255) & ~(size_t)255;
    return p;
  };
  auto cdiv = [](long long a, long long b) -> int { return (int)((a + b - 1) / b); };

  // ping-pong regions
  float* R0 = (float*)A((size_t)NS_ * 16 * L1_ * 4);  // c1 / c2 / c3 / adj
  float* R1 = (float*)A((size_t)NS_ * 16 * L1P_ * 4); // p1 / p2 / f2
  float* bnbuf = (float*)A(6 * 64 * 4);
  float *sc1 = bnbuf, *sh1 = bnbuf + 64, *sc2 = bnbuf + 128, *sh2 = bnbuf + 192,
        *sc3 = bnbuf + 256, *sh3 = bnbuf + 320;
  float*     f1   = (float*)A((size_t)NS_ * 4096 * 4);
  _Float16*  f1h  = (_Float16*)A((size_t)NS_ * 4096 * 2);
  float*     dinv = (float*)A((size_t)NS_ * 64 * 4);
  _Float16*  seqh = (_Float16*)A((size_t)NS_ * 4096 * 2);
  _Float16*  qwh  = (_Float16*)A(4096 * 2);
  _Float16*  kwh  = (_Float16*)A(4096 * 2);
  _Float16*  vwh  = (_Float16*)A(4096 * 2);
  _Float16*  g1h  = (_Float16*)A(4096 * 2);
  _Float16*  g2h  = (_Float16*)A(4096 * 2);
  _Float16*  wih0h = (_Float16*)A((size_t)4096 * 4096 * 2);
  _Float16*  whh0h = (_Float16*)A((size_t)4096 * 1024 * 2);
  _Float16*  wih1h = (_Float16*)A((size_t)4096 * 1024 * 2);
  _Float16*  whh1h = (_Float16*)A((size_t)4096 * 1024 * 2);
  float*     gates = (float*)A((size_t)B_ * 4096 * 4);
  char*      stbase = A(1310720);
  float*     c0s = (float*)stbase;
  float*     c1s = c0s + 65536;
  float*     h0f = c1s + 65536;
  float*     h1f = h0f + 65536;
  _Float16*  h0h = (_Float16*)(h1f + 65536);
  _Float16*  h1h = h0h + 65536;
  _Float16*  y0h = (_Float16*)A((size_t)B_ * T_ * HID_ * 2);

  float* c1 = R0; float* p1 = R1; float* c2 = R0; float* p2 = R1;
  float* c3 = R0; float* adj = R0; float* f2 = R1;

  // weights -> f16
  k_f32h<<<cdiv(4096, 256), 256, 0, stream>>>(q_w, qwh, 4096);
  k_f32h<<<cdiv(4096, 256), 256, 0, stream>>>(k_w, kwh, 4096);
  k_f32h<<<cdiv(4096, 256), 256, 0, stream>>>(v_w, vwh, 4096);
  k_f32h<<<cdiv(4096, 256), 256, 0, stream>>>(g1w, g1h, 4096);
  k_f32h<<<cdiv(4096, 256), 256, 0, stream>>>(g2w, g2h, 4096);
  k_f32h<<<cdiv(16777216, 256), 256, 0, stream>>>(wih0, wih0h, 16777216);
  k_f32h<<<cdiv(4194304, 256), 256, 0, stream>>>(whh0, whh0h, 4194304);
  k_f32h<<<cdiv(4194304, 256), 256, 0, stream>>>(wih1, wih1h, 4194304);
  k_f32h<<<cdiv(4194304, 256), 256, 0, stream>>>(whh1, whh1h, 4194304);

  // conv stack
  int tot1 = NS_ * 16 * L1_;
  k_conv1<<<cdiv(tot1, 256), 256, 0, stream>>>(data, c1w, c1b, c1);
  k_bnstats<<<16, 256, 0, stream>>>(c1, 16, NS_ * L1_, L1_, bn1g, bn1b, sc1, sh1);
  int totp1 = NS_ * 16 * L1P_;
  k_bnpool<<<cdiv(totp1, 256), 256, 0, stream>>>(c1, p1, 16, L1_, L1P_, 8, 4, sc1, sh1, totp1);
  int tot2 = NS_ * 32 * L2_;
  k_conv<<<cdiv(tot2, 256), 256, 0, stream>>>(p1, c2w, c2b, c2, 32, 16, 16, L1P_, L2_, tot2);
  k_bnstats<<<32, 256, 0, stream>>>(c2, 32, NS_ * L2_, L2_, bn2g, bn2b, sc2, sh2);
  int totp2 = NS_ * 32 * L2P_;
  k_bnpool<<<cdiv(totp2, 256), 256, 0, stream>>>(c2, p2, 32, L2_, L2P_, 4, 2, sc2, sh2, totp2);
  int tot3 = NS_ * 64 * L3_;
  k_conv<<<cdiv(tot3, 256), 256, 0, stream>>>(p2, c3w, c3b, c3, 64, 32, 8, L2P_, L3_, tot3);
  k_bnstats<<<64, 256, 0, stream>>>(c3, 64, NS_ * L3_, L3_, bn3g, bn3b, sc3, sh3);
  int totf1 = NS_ * 4096;
  k_f1<<<cdiv(totf1, 256), 256, 0, stream>>>(c3, sc3, sh3, f1, f1h);

  // attention + gaussian adjacency + gumbel
  k_attention<<<NS_, 256, 0, stream>>>(f1, f1h, qwh, q_b, kwh, k_b, vwh, v_b, u, adj);
  k_topk<<<NS_, 256, 0, stream>>>(adj, dinv);

  // GCN x2
  k_gcn<<<NS_, 256, 0, stream>>>(f1, dinv, g1h, g1b, f2, (_Float16*)nullptr);
  k_gcn<<<NS_, 256, 0, stream>>>(f2, dinv, g2h, g2b, (float*)nullptr, seqh);

  // zero LSTM states (deterministic each call)
  k_zero<<<cdiv(327680, 256), 256, 0, stream>>>((unsigned*)stbase, 327680);

  // LSTM layer 0
  for (int t = 0; t < T_; ++t) {
    k_lstm_gemm<<<128, 256, 0, stream>>>(seqh + (size_t)t * 4096, T_ * 4096, 4096,
                                         h0h, wih0h, whh0h, gates);
    k_lstm_cell<<<cdiv(B_ * HID_, 256), 256, 0, stream>>>(gates, bih0, bhh0, c0s, h0h,
                                                          h0f, y0h, t);
  }
  // LSTM layer 1
  for (int t = 0; t < T_; ++t) {
    k_lstm_gemm<<<128, 256, 0, stream>>>(y0h + (size_t)t * HID_, T_ * HID_, 1024,
                                         h1h, wih1h, whh1h, gates);
    k_lstm_cell<<<cdiv(B_ * HID_, 256), 256, 0, stream>>>(gates, bih1, bhh1, c1s, h1h,
                                                          h1f, (_Float16*)nullptr, t);
  }

  // classifier on final hidden state
  k_cls<<<1, 320, 0, stream>>>(h1f, clsw, clsb, out);
}
